// PointerSequenceToSequence_10977936408658
// MI455X (gfx1250) — matrix operand order repaired and checked by reading, not compiled, not run on hardware
//
#include <hip/hip_runtime.h>
#include <hip/hip_bf16.h>
#include <math.h>

// Problem constants (match reference)
#define BB 128
#define LL 64
#define IN 2
#define DD 256
#define HH 512
#define H3 1536
#define XC 768   // D + H (decoder GRU input width)
#define AQ 512   // A_S2S
#define AP 256   // A_PTR
#define NEGV (-1e9f)

typedef __attribute__((ext_vector_type(16))) _Float16 v16h;
typedef __attribute__((ext_vector_type(8)))  _Float16 v8h;
typedef __attribute__((ext_vector_type(8)))  float    v8f;

union V16 { v16h v; v8h h[2]; };

// ---------------------------------------------------------------------------
// Generic f16 WMMA GEMM:  C[M,N] = A[M,K] (row-major, row stride lda)
//                                * W[N,K]^T (weights stored [N,K] row-major)
//                                + bias[N] (optional)
// One wave per block: one 16-row M tile x two 16-col N tiles (A reuse).
// Optional fp32 and fp16 outputs.
// ---------------------------------------------------------------------------
__global__ __launch_bounds__(32)
void wmma_gemm_f16(const _Float16* __restrict__ A, int lda,
                   const _Float16* __restrict__ W,
                   const float* __restrict__ bias,
                   float* __restrict__ Cf, _Float16* __restrict__ Ch,
                   int N, int K)
{
    const int lane = threadIdx.x & 31;
    const int half = lane >> 4;          // 0 or 1
    const int l16  = lane & 15;
    const int n0   = blockIdx.x * 32;    // two adjacent 16-wide N tiles
    const int m0   = blockIdx.y * 16;

    // A fragment: lane l<16 -> row m0+l, K chunks [k..k+7] and [k+16..k+23]
    //             lane l>=16 -> same row, K chunks [k+8..k+15], [k+24..k+31]
    const _Float16* a_ptr = A + (size_t)(m0 + l16) * lda + half * 8;
    // B fragment: lane-per-column, same K chunks as A
    const _Float16* w0 = W + (size_t)(n0 + l16) * K + half * 8;
    const _Float16* w1 = w0 + (size_t)16 * K;

    v8f acc0 = {};
    v8f acc1 = {};
    for (int k = 0; k < K; k += 32) {
        V16 a, b0, b1;
        a.h[0]  = *(const v8h*)(a_ptr + k);
        a.h[1]  = *(const v8h*)(a_ptr + k + 16);
        b0.h[0] = *(const v8h*)(w0 + k);
        b0.h[1] = *(const v8h*)(w0 + k + 16);
        b1.h[0] = *(const v8h*)(w1 + k);
        b1.h[1] = *(const v8h*)(w1 + k + 16);
        __builtin_prefetch(w0 + k + 32, 0, 0);   // global_prefetch_b8
        __builtin_prefetch(w1 + k + 32, 0, 0);
        acc0 = __builtin_amdgcn_wmma_f32_16x16x32_f16(false, a.v, false, b0.v,
                                                      (short)0, acc0, false, false);
        acc1 = __builtin_amdgcn_wmma_f32_16x16x32_f16(false, a.v, false, b1.v,
                                                      (short)0, acc1, false, false);
    }

    // C/D layout: VGPR r -> (M = r + 8*half, N = l16)
    const int m_base = m0 + half * 8;
    {
        const int n_out = n0 + l16;
        const float bv = bias ? bias[n_out] : 0.0f;
        #pragma unroll
        for (int r = 0; r < 8; ++r) {
            const size_t o = (size_t)(m_base + r) * N + n_out;
            const float v = acc0[r] + bv;
            if (Cf) Cf[o] = v;
            if (Ch) Ch[o] = (_Float16)v;
        }
    }
    {
        const int n_out = n0 + 16 + l16;
        const float bv = bias ? bias[n_out] : 0.0f;
        #pragma unroll
        for (int r = 0; r < 8; ++r) {
            const size_t o = (size_t)(m_base + r) * N + n_out;
            const float v = acc1[r] + bv;
            if (Cf) Cf[o] = v;
            if (Ch) Ch[o] = (_Float16)v;
        }
    }
}

// ---------------------------------------------------------------------------
// Weight conversion kernels (fp32 -> fp16, optional transpose to [N,K])
// ---------------------------------------------------------------------------
__global__ void cvt_f16_kernel(const float* __restrict__ src,
                               _Float16* __restrict__ dst, int n)
{
    int i = blockIdx.x * blockDim.x + threadIdx.x;
    if (i < n) dst[i] = (_Float16)src[i];
}

__global__ void cvt_f16_T_kernel(const float* __restrict__ src,  // [R,C]
                                 _Float16* __restrict__ dst,     // [C,R]
                                 int R, int C)
{
    int i = blockIdx.x * blockDim.x + threadIdx.x;
    if (i >= R * C) return;
    int r = i / C, c = i % C;
    dst[(size_t)c * R + r] = (_Float16)src[i];
}

// ---------------------------------------------------------------------------
// dense_inputs = inputs @ W_in^T + b_in   (K=2, trivial)
// ---------------------------------------------------------------------------
__global__ void input_dense_kernel(const float* __restrict__ inputs,  // [B,L,2]
                                   const float* __restrict__ W_in,    // [D,2]
                                   const float* __restrict__ b_in,    // [D]
                                   float* __restrict__ dense_f,
                                   _Float16* __restrict__ dense_h)
{
    int i = blockIdx.x * blockDim.x + threadIdx.x;
    if (i >= BB * LL * DD) return;
    int d = i % DD;
    int bl = i / DD;
    float v = inputs[bl * 2 + 0] * W_in[d * 2 + 0]
            + inputs[bl * 2 + 1] * W_in[d * 2 + 1] + b_in[d];
    dense_f[i] = v;
    dense_h[i] = (_Float16)v;
}

// ---------------------------------------------------------------------------
// State init: encoder/decoder hidden = 0, step_mask = 1, idx = 0
// ---------------------------------------------------------------------------
__global__ void init_state_kernel(float* __restrict__ hEnc_f, _Float16* __restrict__ hEnc_h,
                                  float* __restrict__ hDec_f, _Float16* __restrict__ hDec_h,
                                  float* __restrict__ step_mask, int* __restrict__ idx)
{
    int i = blockIdx.x * blockDim.x + threadIdx.x;
    if (i < BB * HH) {
        hEnc_f[i] = 0.0f; hEnc_h[i] = (_Float16)0.0f;
        hDec_f[i] = 0.0f; hDec_h[i] = (_Float16)0.0f;
    }
    if (i < BB * LL) step_mask[i] = 1.0f;
    if (i < BB) idx[i] = 0;
}

// ---------------------------------------------------------------------------
// GRU gate nonlinearity + state update.
// If enc_out_f != null: also write masked enc_out slice t and capture hidden0.
// ---------------------------------------------------------------------------
__global__ void gru_elem_kernel(const float* __restrict__ gi, const float* __restrict__ gh,
                                const float* __restrict__ bi, const float* __restrict__ bh,
                                const float* __restrict__ h_old,
                                float* __restrict__ h_new, _Float16* __restrict__ h_new_h,
                                float* __restrict__ enc_out_f, _Float16* __restrict__ enc_out_h,
                                float* __restrict__ hid0_f, _Float16* __restrict__ hid0_h,
                                const int* __restrict__ lengths, int t)
{
    int i = blockIdx.x * blockDim.x + threadIdx.x;
    if (i >= BB * HH) return;
    int b = i / HH, j = i % HH;
    size_t base = (size_t)b * H3 + j;
    float ir = gi[base]          + bi[j];
    float iz = gi[base + HH]     + bi[j + HH];
    float inn = gi[base + 2*HH]  + bi[j + 2*HH];
    float hr = gh[base]          + bh[j];
    float hz = gh[base + HH]     + bh[j + HH];
    float hn = gh[base + 2*HH]   + bh[j + 2*HH];
    float r = 1.0f / (1.0f + __expf(-(ir + hr)));
    float z = 1.0f / (1.0f + __expf(-(iz + hz)));
    float n = tanhf(inn + r * hn);
    float hv = (1.0f - z) * n + z * h_old[i];
    h_new[i] = hv;
    h_new_h[i] = (_Float16)hv;
    if (enc_out_f) {
        int len = lengths[b];
        float m = (t < len) ? hv : 0.0f;
        size_t eo = ((size_t)b * LL + t) * HH + j;
        enc_out_f[eo] = m;
        enc_out_h[eo] = (_Float16)m;
        if (t == len - 1) { hid0_f[i] = hv; hid0_h[i] = (_Float16)hv; }
    }
}

// ---------------------------------------------------------------------------
// Bahdanau seq2seq attention for one decoder step:
//   scores[b,l] = v . tanh(q[b,:] + enc_keys[b,l,:]); masked softmax;
//   context[b,:] = sum_l w[l]*enc_out[b,l,:]; x_cat = [context | dense[b,idx[b]]]
// One block per batch row, 64 threads (thread == l).
// ---------------------------------------------------------------------------
__global__ __launch_bounds__(64)
void s2s_attn_kernel(const float* __restrict__ q,         // [B,AQ]
                     const float* __restrict__ enc_keys,  // [B,L,AQ]
                     const float* __restrict__ s2s_v,     // [AQ]
                     const float* __restrict__ enc_out_f, // [B,L,H]
                     const _Float16* __restrict__ dense_h,// [B,L,D]
                     const int* __restrict__ lengths,
                     const int* __restrict__ idx,
                     _Float16* __restrict__ x_cat,        // [B, H+D]
                     int t)
{
    int b = blockIdx.x;
    int l = threadIdx.x;
    __shared__ float w[LL];

    const float* qb = q + (size_t)b * AQ;
    const float* kb = enc_keys + ((size_t)b * LL + l) * AQ;
    float s = 0.0f;
    for (int a = 0; a < AQ; ++a)
        s += tanhf(qb[a] + kb[a]) * s2s_v[a];
    int len = lengths[b];
    bool valid = (t < len) && (l < len);
    w[l] = valid ? s : NEGV;
    __syncthreads();
    if (l == 0) {
        float mx = -3.0e38f;
        for (int i = 0; i < LL; ++i) mx = fmaxf(mx, w[i]);
        float sum = 0.0f;
        for (int i = 0; i < LL; ++i) { float e = __expf(w[i] - mx); w[i] = e; sum += e; }
        float inv = 1.0f / sum;
        for (int i = 0; i < LL; ++i) w[i] *= inv;
    }
    __syncthreads();
    // context: each thread owns 8 hidden units
    for (int jj = 0; jj < 8; ++jj) {
        int j = l * 8 + jj;
        float c = 0.0f;
        for (int i = 0; i < LL; ++i)
            c += w[i] * enc_out_f[((size_t)b * LL + i) * HH + j];
        x_cat[(size_t)b * XC + j] = (_Float16)c;
    }
    // dec_in = dense_inputs[b, idx[b], :]
    int id = idx[b];
    for (int d = l; d < DD; d += LL)
        x_cat[(size_t)b * XC + HH + d] = dense_h[((size_t)b * LL + id) * DD + d];
}

// ---------------------------------------------------------------------------
// Pointer attention + argmax + teacher-forced state update for one step.
// One block per batch row, 64 threads (thread == l).
// ---------------------------------------------------------------------------
__global__ __launch_bounds__(64)
void ptr_attn_kernel(const float* __restrict__ pq,        // [B,AP]
                     const float* __restrict__ ptr_keys,  // [B,L,AP]
                     const float* __restrict__ ptr_v,     // [AP]
                     const int* __restrict__ lengths,
                     const int* __restrict__ targets,     // [B,L]
                     float* __restrict__ logits,          // [B,L,L]
                     float* __restrict__ preds,           // [B,L] (as float)
                     int* __restrict__ idx,
                     float* __restrict__ step_mask,       // [B,L]
                     int t)
{
    int b = blockIdx.x;
    int l = threadIdx.x;
    __shared__ float sc[LL];

    const float* pb = pq + (size_t)b * AP;
    const float* kb = ptr_keys + ((size_t)b * LL + l) * AP;
    float s = 0.0f;
    for (int a = 0; a < AP; ++a)
        s += tanhf(pb[a] + kb[a]) * ptr_v[a];
    int len = lengths[b];
    bool pm = (step_mask[(size_t)b * LL + l] > 0.0f) && (t < len) && (l < len);
    float v = pm ? s : NEGV;
    sc[l] = v;
    logits[((size_t)b * LL + t) * LL + l] = v;
    __syncthreads();
    if (l == 0) {
        int best = 0; float bv = sc[0];
        for (int i = 1; i < LL; ++i) if (sc[i] > bv) { bv = sc[i]; best = i; }
        preds[(size_t)b * LL + t] = (float)best;
        int tgt = targets[(size_t)b * LL + t];
        idx[b] = tgt;                       // teacher forcing
        step_mask[(size_t)b * LL + tgt] = 0.0f;
    }
}

// ---------------------------------------------------------------------------
// Host orchestration
// ---------------------------------------------------------------------------
static inline void gemm(const _Float16* A, int lda, const _Float16* W,
                        const float* bias, float* Cf, _Float16* Ch,
                        int M, int N, int K, hipStream_t s)
{
    dim3 grid(N / 32, M / 16);
    wmma_gemm_f16<<<grid, dim3(32), 0, s>>>(A, lda, W, bias, Cf, Ch, N, K);
}

extern "C" void kernel_launch(void* const* d_in, const int* in_sizes, int n_in,
                              void* d_out, int out_size, void* d_ws, size_t ws_size,
                              hipStream_t stream)
{
    (void)in_sizes; (void)n_in; (void)out_size; (void)ws_size;

    // ---- inputs (setup_inputs dict order) ----
    const float* inputs  = (const float*)d_in[0];
    const int*   lengths = (const int*)  d_in[1];
    const int*   targets = (const int*)  d_in[2];
    const float* W_in    = (const float*)d_in[3];
    const float* b_in    = (const float*)d_in[4];
    const float* enc_Wi  = (const float*)d_in[5];
    const float* enc_Wh  = (const float*)d_in[6];
    const float* enc_bi  = (const float*)d_in[7];
    const float* enc_bh  = (const float*)d_in[8];
    const float* dec_Wi  = (const float*)d_in[9];
    const float* dec_Wh  = (const float*)d_in[10];
    const float* dec_bi  = (const float*)d_in[11];
    const float* dec_bh  = (const float*)d_in[12];
    const float* s2s_Wq  = (const float*)d_in[13];
    const float* s2s_Wk  = (const float*)d_in[14];
    const float* s2s_v   = (const float*)d_in[15];
    const float* W_out   = (const float*)d_in[16];
    const float* b_out   = (const float*)d_in[17];
    const float* ptr_Wq  = (const float*)d_in[18];
    const float* ptr_Wk  = (const float*)d_in[19];
    const float* ptr_v   = (const float*)d_in[20];

    float* logits = (float*)d_out;                       // [B,L,L]
    float* preds  = logits + (size_t)BB * LL * LL;       // [B,L]

    // ---- workspace carve-up (256B aligned) ----
    char* p = (char*)d_ws;
    auto alloc = [&](size_t elems, size_t esz) -> void* {
        void* r = (void*)p;
        p += (elems * esz + 255) & ~(size_t)255;
        return r;
    };
    // f16 weights ([N,K] layout for the GEMM)
    _Float16* encWi_h = (_Float16*)alloc((size_t)H3 * DD, 2);
    _Float16* encWh_h = (_Float16*)alloc((size_t)H3 * HH, 2);
    _Float16* decWi_h = (_Float16*)alloc((size_t)H3 * XC, 2);
    _Float16* decWh_h = (_Float16*)alloc((size_t)H3 * HH, 2);
    _Float16* WqT_h   = (_Float16*)alloc((size_t)AQ * HH, 2);
    _Float16* WkT_h   = (_Float16*)alloc((size_t)AQ * HH, 2);
    _Float16* Wout_h  = (_Float16*)alloc((size_t)DD * HH, 2);
    _Float16* pWqT_h  = (_Float16*)alloc((size_t)AP * DD, 2);
    _Float16* pWkT_h  = (_Float16*)alloc((size_t)AP * DD, 2);
    // f16 activations
    _Float16* dense_h  = (_Float16*)alloc((size_t)BB * LL * DD, 2);
    _Float16* encout_h = (_Float16*)alloc((size_t)BB * LL * HH, 2);
    _Float16* hEnc_h   = (_Float16*)alloc((size_t)BB * HH, 2);
    _Float16* hDec_h   = (_Float16*)alloc((size_t)BB * HH, 2);
    _Float16* xcat_h   = (_Float16*)alloc((size_t)BB * XC, 2);
    _Float16* dout_h   = (_Float16*)alloc((size_t)BB * DD, 2);
    // f32 activations
    float* dense_f  = (float*)alloc((size_t)BB * LL * DD, 4);
    float* encout_f = (float*)alloc((size_t)BB * LL * HH, 4);
    float* enckeys  = (float*)alloc((size_t)BB * LL * AQ, 4);
    float* ptrkeys  = (float*)alloc((size_t)BB * LL * AP, 4);
    float* hEnc_f   = (float*)alloc((size_t)BB * HH, 4);
    float* hDec_f   = (float*)alloc((size_t)BB * HH, 4);
    float* gi       = (float*)alloc((size_t)BB * H3, 4);
    float* gh       = (float*)alloc((size_t)BB * H3, 4);
    float* q_f      = (float*)alloc((size_t)BB * AQ, 4);
    float* dout_f   = (float*)alloc((size_t)BB * DD, 4);
    float* pq_f     = (float*)alloc((size_t)BB * AP, 4);
    float* smask    = (float*)alloc((size_t)BB * LL, 4);
    int*   idx      = (int*)  alloc((size_t)BB, 4);

    const int TPB = 256;
    auto blocks = [](size_t n, int t) { return (unsigned)((n + t - 1) / t); };

    // ---- 1) weight conversion (fp32 -> fp16 [N,K]) ----
    cvt_f16_kernel<<<blocks((size_t)H3 * DD, TPB), TPB, 0, stream>>>(enc_Wi, encWi_h, H3 * DD);
    cvt_f16_kernel<<<blocks((size_t)H3 * HH, TPB), TPB, 0, stream>>>(enc_Wh, encWh_h, H3 * HH);
    cvt_f16_kernel<<<blocks((size_t)H3 * XC, TPB), TPB, 0, stream>>>(dec_Wi, decWi_h, H3 * XC);
    cvt_f16_kernel<<<blocks((size_t)H3 * HH, TPB), TPB, 0, stream>>>(dec_Wh, decWh_h, H3 * HH);
    cvt_f16_kernel<<<blocks((size_t)DD * HH, TPB), TPB, 0, stream>>>(W_out, Wout_h, DD * HH);
    cvt_f16_T_kernel<<<blocks((size_t)HH * AQ, TPB), TPB, 0, stream>>>(s2s_Wq, WqT_h, HH, AQ);
    cvt_f16_T_kernel<<<blocks((size_t)HH * AQ, TPB), TPB, 0, stream>>>(s2s_Wk, WkT_h, HH, AQ);
    cvt_f16_T_kernel<<<blocks((size_t)DD * AP, TPB), TPB, 0, stream>>>(ptr_Wq, pWqT_h, DD, AP);
    cvt_f16_T_kernel<<<blocks((size_t)DD * AP, TPB), TPB, 0, stream>>>(ptr_Wk, pWkT_h, DD, AP);

    // ---- 2) init state + input dense ----
    init_state_kernel<<<blocks((size_t)BB * HH, TPB), TPB, 0, stream>>>(
        hEnc_f, hEnc_h, hDec_f, hDec_h, smask, idx);
    input_dense_kernel<<<blocks((size_t)BB * LL * DD, TPB), TPB, 0, stream>>>(
        inputs, W_in, b_in, dense_f, dense_h);

    // ---- 3) ptr_keys = dense_inputs @ ptr_Wk  (M=8192, N=256, K=256) ----
    gemm(dense_h, DD, pWkT_h, nullptr, ptrkeys, nullptr, BB * LL, AP, DD, stream);

    // ---- 4) encoder GRU over time ----
    for (int t = 0; t < LL; ++t) {
        // gi = x_t @ enc_Wi^T  (A rows strided through [B,L,D])
        gemm(dense_h + (size_t)t * DD, LL * DD, encWi_h, nullptr, gi, nullptr,
             BB, H3, DD, stream);
        // gh = h @ enc_Wh^T
        gemm(hEnc_h, HH, encWh_h, nullptr, gh, nullptr, BB, H3, HH, stream);
        gru_elem_kernel<<<blocks((size_t)BB * HH, TPB), TPB, 0, stream>>>(
            gi, gh, enc_bi, enc_bh, hEnc_f, hEnc_f, hEnc_h,
            encout_f, encout_h, hDec_f, hDec_h, lengths, t);
    }

    // ---- 5) enc_keys = enc_out @ s2s_Wk  (M=8192, N=512, K=512) ----
    gemm(encout_h, HH, WkT_h, nullptr, enckeys, nullptr, BB * LL, AQ, HH, stream);

    // ---- 6) decoder loop ----
    for (int t = 0; t < LL; ++t) {
        // q = h @ s2s_Wq
        gemm(hDec_h, HH, WqT_h, nullptr, q_f, nullptr, BB, AQ, HH, stream);
        // attention + x_cat = [context | dense[b, idx[b]]]
        s2s_attn_kernel<<<BB, LL, 0, stream>>>(
            q_f, enckeys, s2s_v, encout_f, dense_h, lengths, idx, xcat_h, t);
        // decoder GRU
        gemm(xcat_h, XC, decWi_h, nullptr, gi, nullptr, BB, H3, XC, stream);
        gemm(hDec_h, HH, decWh_h, nullptr, gh, nullptr, BB, H3, HH, stream);
        gru_elem_kernel<<<blocks((size_t)BB * HH, TPB), TPB, 0, stream>>>(
            gi, gh, dec_bi, dec_bh, hDec_f, hDec_f, hDec_h,
            nullptr, nullptr, nullptr, nullptr, lengths, t);
        // dense_out = h_new @ W_out^T + b_out
        gemm(hDec_h, HH, Wout_h, b_out, dout_f, dout_h, BB, DD, HH, stream);
        // pq = dense_out @ ptr_Wq
        gemm(dout_h, DD, pWqT_h, nullptr, pq_f, nullptr, BB, AP, DD, stream);
        // pointer attention, logits/preds, teacher-forced state update
        ptr_attn_kernel<<<BB, LL, 0, stream>>>(
            pq_f, ptrkeys, ptr_v, lengths, targets, logits, preds, idx, smask, t);
    }
}
